// PAMoE_4105988735153
// MI455X (gfx1250) — matrix operand order repaired and checked by reading, not compiled, not run on hardware
//
#include <hip/hip_runtime.h>
#include <hip/hip_bf16.h>

#define B_NUM 4
#define S_NUM 4096
#define D_DIM 1024
#define E_NUM 8
#define F_DIM 2048
#define K_CAP 512
#define ROWS  (B_NUM * E_NUM * K_CAP)   // 16384 dispatched rows

typedef __bf16 bf16;
typedef __bf16 v8bf  __attribute__((ext_vector_type(8)));
typedef __bf16 v16bf __attribute__((ext_vector_type(16)));
typedef float  v8f   __attribute__((ext_vector_type(8)));

#define LDK 72   // padded LDS row stride (bf16 elems) for 64-deep K chunks

#define CAT16(lo, hi) __builtin_shufflevector(lo, hi, 0,1,2,3,4,5,6,7,8,9,10,11,12,13,14,15)

// Branch-free tanh-form GELU: 0.5x(1+tanh(0.79788456(x+0.044715x^3)))
//   = x * (1 - 1/(exp(2y)+1));  error << bf16 quantization of h.
__device__ __forceinline__ float gelu_f(float x) {
    float x3 = x * x * x;
    float y2 = 1.5957691216f * x + 0.0713548163f * x3;   // 2*y
    float t  = __expf(y2);
    return x * (1.f - __builtin_amdgcn_rcpf(t + 1.f));
}

// ---------------------------------------------------------------------------
// 1) f32 -> bf16 straight convert (inputs)
// ---------------------------------------------------------------------------
__global__ __launch_bounds__(256)
void cvt_inputs_kernel(const float* __restrict__ in, bf16* __restrict__ out) {
    size_t i = ((size_t)blockIdx.x * 256 + threadIdx.x) * 8;
    float4 a = *(const float4*)(in + i);
    float4 b = *(const float4*)(in + i + 4);
    v8bf o;
    o[0] = (bf16)a.x; o[1] = (bf16)a.y; o[2] = (bf16)a.z; o[3] = (bf16)a.w;
    o[4] = (bf16)b.x; o[5] = (bf16)b.y; o[6] = (bf16)b.z; o[7] = (bf16)b.w;
    *(v8bf*)(out + i) = o;
}

// ---------------------------------------------------------------------------
// 2) f32 [R][C] -> bf16 [C][R] tiled transpose-convert (expert = grid.z),
//    optional per-source-row scale (folds gamma into W2).
// ---------------------------------------------------------------------------
__global__ __launch_bounds__(256)
void transpose_cvt_kernel(const float* __restrict__ src, bf16* __restrict__ dst,
                          int R, int C, const float* __restrict__ rowscale) {
    __shared__ float tile[32][33];
    size_t eoff = (size_t)blockIdx.z * R * C;
    int c0 = blockIdx.x * 32, r0 = blockIdx.y * 32;
    int tx = threadIdx.x, ty = threadIdx.y;          // (32, 8)
    #pragma unroll
    for (int i = 0; i < 32; i += 8) {
        float v = src[eoff + (size_t)(r0 + ty + i) * C + c0 + tx];
        if (rowscale) v *= rowscale[(size_t)blockIdx.z * R + r0 + ty + i];
        tile[ty + i][tx] = v;
    }
    __syncthreads();
    #pragma unroll
    for (int i = 0; i < 32; i += 8)
        dst[eoff + (size_t)(c0 + ty + i) * R + r0 + tx] = (bf16)tile[tx][ty + i];
}

// ---------------------------------------------------------------------------
// 3) Router: logits + softmax -> probsT [B][E][S].  One wave per token.
// ---------------------------------------------------------------------------
__global__ __launch_bounds__(256)
void router_kernel(const float* __restrict__ inputs, const float* __restrict__ Wr,
                   float* __restrict__ logits_out, float* __restrict__ probsT) {
    int wid  = threadIdx.x >> 5;
    int lane = threadIdx.x & 31;
    int token = blockIdx.x * 8 + wid;                // 16384 tokens
    int b = token >> 12, s = token & 4095;
    const float* xr = inputs + (size_t)token * D_DIM;

    float acc[E_NUM];
    #pragma unroll
    for (int e = 0; e < E_NUM; ++e) acc[e] = 0.f;

    for (int d = lane; d < D_DIM; d += 32) {
        float x = xr[d];
        const float* wr = Wr + d * E_NUM;
        #pragma unroll
        for (int e = 0; e < E_NUM; ++e) acc[e] += x * wr[e];
    }
    #pragma unroll
    for (int e = 0; e < E_NUM; ++e)
        #pragma unroll
        for (int m = 16; m > 0; m >>= 1) acc[e] += __shfl_xor(acc[e], m, 32);

    if (lane == 0) {
        float mx = acc[0];
        #pragma unroll
        for (int e = 1; e < E_NUM; ++e) mx = fmaxf(mx, acc[e]);
        float ex[E_NUM], sum = 0.f;
        #pragma unroll
        for (int e = 0; e < E_NUM; ++e) { ex[e] = expf(acc[e] - mx); sum += ex[e]; }
        float inv = 1.f / sum;
        #pragma unroll
        for (int e = 0; e < E_NUM; ++e) {
            logits_out[(size_t)token * E_NUM + e] = acc[e];
            probsT[((size_t)(b * E_NUM + e)) * S_NUM + s] = ex[e] * inv;
        }
    }
}

// ---------------------------------------------------------------------------
// 4) Expert-choice top-512 of 4096 via LDS bitonic sort (descending).
// ---------------------------------------------------------------------------
__global__ __launch_bounds__(512)
void topk_kernel(const float* __restrict__ probsT, int* __restrict__ idxbuf,
                 float* __restrict__ wgtbuf) {
    __shared__ unsigned long long keys[S_NUM];
    int be = blockIdx.x;                              // b*E + e, 32 blocks
    const float* p = probsT + (size_t)be * S_NUM;
    for (int i = threadIdx.x; i < S_NUM; i += 512)
        keys[i] = ((unsigned long long)__float_as_uint(p[i]) << 32) | (unsigned)i;
    __syncthreads();

    for (int k = 2; k <= S_NUM; k <<= 1) {
        for (int j = k >> 1; j > 0; j >>= 1) {
            for (int i = threadIdx.x; i < S_NUM; i += 512) {
                int l = i ^ j;
                if (l > i) {
                    unsigned long long a = keys[i], c = keys[l];
                    bool up = (i & k) == 0;           // descending overall
                    if (up ? (a < c) : (a > c)) { keys[i] = c; keys[l] = a; }
                }
            }
            __syncthreads();
        }
    }
    unsigned long long kv = keys[threadIdx.x];        // top 512, descending
    wgtbuf[(size_t)be * K_CAP + threadIdx.x] = __uint_as_float((unsigned)(kv >> 32));
    idxbuf[(size_t)be * K_CAP + threadIdx.x] = (int)(kv & 0xffffffffu);
}

// ---------------------------------------------------------------------------
// 5) gdot[e,n] = sum_f gamma[e,f]*W2[e,f,n]; bdot likewise for beta.
//    Deterministic two-pass (partials over 16 f-chunks, then reduce).
// ---------------------------------------------------------------------------
__global__ __launch_bounds__(256)
void w2dot_partial_kernel(const float* __restrict__ W2, const float* __restrict__ gamma,
                          const float* __restrict__ beta, float* __restrict__ gpart,
                          float* __restrict__ bpart) {
    int nc = blockIdx.x;                      // 4
    int fc = blockIdx.y;                      // 16
    int e  = blockIdx.z;                      // 8
    int n  = nc * 256 + threadIdx.x;
    const float* w  = W2 + ((size_t)e * F_DIM + fc * 128) * D_DIM + n;
    const float* g  = gamma + (size_t)e * F_DIM + fc * 128;
    const float* bt = beta  + (size_t)e * F_DIM + fc * 128;
    float gs = 0.f, bs = 0.f;
    for (int f = 0; f < 128; ++f) {
        float wv = w[(size_t)f * D_DIM];
        gs += g[f] * wv;
        bs += bt[f] * wv;
    }
    size_t o = ((size_t)e * D_DIM + n) * 16 + fc;
    gpart[o] = gs;
    bpart[o] = bs;
}

__global__ __launch_bounds__(256)
void w2dot_reduce_kernel(const float* __restrict__ gpart, const float* __restrict__ bpart,
                         float* __restrict__ gdot, float* __restrict__ bdot) {
    int i = blockIdx.x * 256 + threadIdx.x;   // 8192
    float gs = 0.f, bs = 0.f;
    #pragma unroll
    for (int fc = 0; fc < 16; ++fc) { gs += gpart[(size_t)i * 16 + fc]; bs += bpart[(size_t)i * 16 + fc]; }
    gdot[i] = gs;
    bdot[i] = bs;
}

// ---------------------------------------------------------------------------
// 6) fc1: h = gelu(gather(X) @ W1[e] + b1), bf16 out.
//    BM=128 BN=128 BK=64; 8 waves, register double-buffered staging.
// ---------------------------------------------------------------------------
__global__ __launch_bounds__(256)
void fc1_kernel(const bf16* __restrict__ Xbf, const bf16* __restrict__ W1T,
                const float* __restrict__ b1, const int* __restrict__ idxbuf,
                bf16* __restrict__ hbuf) {
    __shared__ __align__(16) bf16 As[128 * LDK];
    __shared__ __align__(16) bf16 Bs[128 * LDK];
    __shared__ unsigned offs[128];

    const int tid = threadIdx.x;
    const int n0 = blockIdx.x * 128;
    const int r0 = blockIdx.y * 128;
    const int e  = (r0 >> 9) & 7;
    const int b  = r0 >> 12;

    if (tid < 128) {
        int r = r0 + tid;
        int k = r & (K_CAP - 1);
        int t = idxbuf[((b * E_NUM + e) << 9) + k];
        offs[tid] = (unsigned)((b * S_NUM + t) * D_DIM);
    }
    __syncthreads();

    const int wid = tid >> 5, lane = tid & 31;
    const int wm = wid & 3, wn = wid >> 2;
    const int lr = lane & 15;
    const bool hihalf = lane >= 16;
    const int srow = tid >> 3;                 // staging row base (0..31)
    const int seg  = tid & 7;                  // staging 8-elem segment

    v8f acc[2][4];
    const v8f vz = {0.f, 0.f, 0.f, 0.f, 0.f, 0.f, 0.f, 0.f};
    #pragma unroll
    for (int t = 0; t < 2; ++t)
        #pragma unroll
        for (int c = 0; c < 4; ++c) acc[t][c] = vz;

    const bf16* W1e = W1T + (size_t)e * F_DIM * D_DIM;   // [F][D]

    v8bf pa[4], pb[4];
    #pragma unroll
    for (int it = 0; it < 4; ++it) {
        pa[it] = *(const v8bf*)(Xbf + offs[srow + it * 32] + seg * 8);
        pb[it] = *(const v8bf*)(W1e + (size_t)(n0 + srow + it * 32) * D_DIM + seg * 8);
    }

    for (int kk = 0; kk < D_DIM; kk += 64) {
        __syncthreads();
        #pragma unroll
        for (int it = 0; it < 4; ++it) {
            *(v8bf*)(&As[(srow + it * 32) * LDK + seg * 8]) = pa[it];
            *(v8bf*)(&Bs[(srow + it * 32) * LDK + seg * 8]) = pb[it];
        }
        __syncthreads();
        if (kk + 64 < D_DIM) {
            #pragma unroll
            for (int it = 0; it < 4; ++it) {
                pa[it] = *(const v8bf*)(Xbf + offs[srow + it * 32] + kk + 64 + seg * 8);
                pb[it] = *(const v8bf*)(W1e + (size_t)(n0 + srow + it * 32) * D_DIM + kk + 64 + seg * 8);
            }
        }
        #pragma unroll
        for (int s = 0; s < 2; ++s) {
            const int akoff = s * 32 + (hihalf ? 8 : 0);
            const int bkoff = s * 32 + (hihalf ? 8 : 0);
            v16bf af[2], bfr[4];
            #pragma unroll
            for (int t = 0; t < 2; ++t) {
                int row = wm * 32 + t * 16 + lr;
                v8bf lo = *(const v8bf*)(&As[row * LDK + akoff]);
                v8bf hi = *(const v8bf*)(&As[row * LDK + akoff + 16]);
                af[t] = CAT16(lo, hi);
            }
            #pragma unroll
            for (int c = 0; c < 4; ++c) {
                int n = wn * 64 + c * 16 + lr;
                v8bf lo = *(const v8bf*)(&Bs[n * LDK + bkoff]);
                v8bf hi = *(const v8bf*)(&Bs[n * LDK + bkoff + 16]);
                bfr[c] = CAT16(lo, hi);
            }
            #pragma unroll
            for (int t = 0; t < 2; ++t)
                #pragma unroll
                for (int c = 0; c < 4; ++c)
                    acc[t][c] = __builtin_amdgcn_wmma_f32_16x16x32_bf16(
                        false, af[t], false, bfr[c], (short)0, acc[t][c], false, false);
        }
    }

    // epilogue: +b1, GELU, bf16 store
    #pragma unroll
    for (int t = 0; t < 2; ++t)
        #pragma unroll
        for (int c = 0; c < 4; ++c) {
            int col = n0 + wn * 64 + c * 16 + lr;
            float bb = b1[e * F_DIM + col];
            #pragma unroll
            for (int i = 0; i < 8; ++i) {
                int row = r0 + wm * 32 + t * 16 + (hihalf ? i + 8 : i);
                hbuf[(size_t)row * F_DIM + col] = (bf16)gelu_f(acc[t][c][i] + bb);
            }
        }
}

// ---------------------------------------------------------------------------
// 7) LayerNorm stats over bf16 h: one wave per row -> (mu, rsqrt(var+eps))
// ---------------------------------------------------------------------------
__global__ __launch_bounds__(256)
void ln_stats_kernel(const bf16* __restrict__ hbuf, float* __restrict__ stats) {
    int wid = threadIdx.x >> 5, lane = threadIdx.x & 31;
    int row = blockIdx.x * 8 + wid;                   // 16384 rows
    const bf16* hp = hbuf + (size_t)row * F_DIM;
    float s = 0.f, s2 = 0.f;
    for (int i = lane * 8; i < F_DIM; i += 32 * 8) {
        v8bf v = *(const v8bf*)(hp + i);
        #pragma unroll
        for (int j = 0; j < 8; ++j) { float f = (float)v[j]; s += f; s2 += f * f; }
    }
    #pragma unroll
    for (int m = 16; m > 0; m >>= 1) { s += __shfl_xor(s, m, 32); s2 += __shfl_xor(s2, m, 32); }
    if (lane == 0) {
        float mu  = s * (1.f / F_DIM);
        float var = fmaxf(s2 * (1.f / F_DIM) - mu * mu, 0.f);
        stats[2 * row]     = mu;
        stats[2 * row + 1] = rsqrtf(var + 1e-5f);
    }
}

// ---------------------------------------------------------------------------
// 8) fc2 (LN folded out): acc = h @ (gamma.*W2)[e];
//    out = inv*acc - inv*mu*gdot + bdot + b2; weighted atomic scatter-add.
// ---------------------------------------------------------------------------
__global__ __launch_bounds__(256)
void fc2_kernel(const bf16* __restrict__ hbuf, const bf16* __restrict__ W2gT,
                const float* __restrict__ gdot, const float* __restrict__ bdot,
                const float* __restrict__ b2, const float* __restrict__ stats,
                const int* __restrict__ idxbuf, const float* __restrict__ wgtbuf,
                float* __restrict__ results) {
    __shared__ __align__(16) bf16 As[128 * LDK];
    __shared__ __align__(16) bf16 Bs[128 * LDK];
    __shared__ float muS[128], invS[128], wS[128];
    __shared__ int   tokS[128];

    const int tid = threadIdx.x;
    const int n0 = blockIdx.x * 128;
    const int r0 = blockIdx.y * 128;
    const int e  = (r0 >> 9) & 7;
    const int b  = r0 >> 12;

    if (tid < 128) {
        int r = r0 + tid;
        int k = r & (K_CAP - 1);
        muS[tid]  = stats[2 * r];
        invS[tid] = stats[2 * r + 1];
        tokS[tid] = idxbuf[((b * E_NUM + e) << 9) + k];
        wS[tid]   = wgtbuf[((b * E_NUM + e) << 9) + k];
    }

    const int wid = tid >> 5, lane = tid & 31;
    const int wm = wid & 3, wn = wid >> 2;
    const int lr = lane & 15;
    const bool hihalf = lane >= 16;
    const int srow = tid >> 3;
    const int seg  = tid & 7;

    v8f acc[2][4];
    const v8f vz = {0.f, 0.f, 0.f, 0.f, 0.f, 0.f, 0.f, 0.f};
    #pragma unroll
    for (int t = 0; t < 2; ++t)
        #pragma unroll
        for (int c = 0; c < 4; ++c) acc[t][c] = vz;

    const bf16* W2e = W2gT + (size_t)e * D_DIM * F_DIM;   // [D][F], gamma folded

    v8bf pa[4], pb[4];
    #pragma unroll
    for (int it = 0; it < 4; ++it) {
        pa[it] = *(const v8bf*)(hbuf + (size_t)(r0 + srow + it * 32) * F_DIM + seg * 8);
        pb[it] = *(const v8bf*)(W2e + (size_t)(n0 + srow + it * 32) * F_DIM + seg * 8);
    }

    for (int kk = 0; kk < F_DIM; kk += 64) {
        __syncthreads();
        #pragma unroll
        for (int it = 0; it < 4; ++it) {
            *(v8bf*)(&As[(srow + it * 32) * LDK + seg * 8]) = pa[it];
            *(v8bf*)(&Bs[(srow + it * 32) * LDK + seg * 8]) = pb[it];
        }
        __syncthreads();
        if (kk + 64 < F_DIM) {
            #pragma unroll
            for (int it = 0; it < 4; ++it) {
                pa[it] = *(const v8bf*)(hbuf + (size_t)(r0 + srow + it * 32) * F_DIM + kk + 64 + seg * 8);
                pb[it] = *(const v8bf*)(W2e + (size_t)(n0 + srow + it * 32) * F_DIM + kk + 64 + seg * 8);
            }
        }
        #pragma unroll
        for (int s = 0; s < 2; ++s) {
            const int akoff = s * 32 + (hihalf ? 8 : 0);
            const int bkoff = s * 32 + (hihalf ? 8 : 0);
            v16bf af[2], bfr[4];
            #pragma unroll
            for (int t = 0; t < 2; ++t) {
                int row = wm * 32 + t * 16 + lr;
                v8bf lo = *(const v8bf*)(&As[row * LDK + akoff]);
                v8bf hi = *(const v8bf*)(&As[row * LDK + akoff + 16]);
                af[t] = CAT16(lo, hi);
            }
            #pragma unroll
            for (int c = 0; c < 4; ++c) {
                int n = wn * 64 + c * 16 + lr;
                v8bf lo = *(const v8bf*)(&Bs[n * LDK + bkoff]);
                v8bf hi = *(const v8bf*)(&Bs[n * LDK + bkoff + 16]);
                bfr[c] = CAT16(lo, hi);
            }
            #pragma unroll
            for (int t = 0; t < 2; ++t)
                #pragma unroll
                for (int c = 0; c < 4; ++c)
                    acc[t][c] = __builtin_amdgcn_wmma_f32_16x16x32_bf16(
                        false, af[t], false, bfr[c], (short)0, acc[t][c], false, false);
        }
    }

    // epilogue: un-fold LN, +b2, weight*val, atomic scatter-add
    #pragma unroll
    for (int t = 0; t < 2; ++t)
        #pragma unroll
        for (int c = 0; c < 4; ++c) {
            int col = n0 + wn * 64 + c * 16 + lr;
            float gd = gdot[e * D_DIM + col];
            float bd = bdot[e * D_DIM + col] + b2[e * D_DIM + col];
            #pragma unroll
            for (int i = 0; i < 8; ++i) {
                int rl = wm * 32 + t * 16 + (hihalf ? i + 8 : i);
                float a = invS[rl];
                float val = a * acc[t][c][i] - a * muS[rl] * gd + bd;
                atomicAdd(&results[((size_t)b * S_NUM + tokS[rl]) * D_DIM + col],
                          wS[rl] * val);
            }
        }
}

// ---------------------------------------------------------------------------
extern "C" void kernel_launch(void* const* d_in, const int* in_sizes, int n_in,
                              void* d_out, int out_size, void* d_ws, size_t ws_size,
                              hipStream_t stream) {
    const float* inputs = (const float*)d_in[0];
    const float* Wr     = (const float*)d_in[1];
    const float* W1     = (const float*)d_in[2];
    const float* b1     = (const float*)d_in[3];
    const float* gamma  = (const float*)d_in[4];
    const float* beta   = (const float*)d_in[5];
    const float* W2     = (const float*)d_in[6];
    const float* b2     = (const float*)d_in[7];

    float* results = (float*)d_out;                               // [B,S,D]
    float* logits  = results + (size_t)B_NUM * S_NUM * D_DIM;     // [B,S,E]

    // workspace layout (bytes)
    char* ws = (char*)d_ws;
    bf16*  Xbf    = (bf16*)(ws);                                  // 32 MB
    bf16*  W1T    = (bf16*)(ws + (size_t)33554432);               // 32 MB  [E][F][D]
    bf16*  W2gT   = (bf16*)(ws + (size_t)67108864);               // 32 MB  [E][D][F] (gamma folded)
    bf16*  hbuf   = (bf16*)(ws + (size_t)100663296);              // 64 MB  [ROWS][F]
    float* probsT = (float*)(ws + (size_t)167772160);             // 0.5 MB [B][E][S]
    float* stats  = (float*)(ws + (size_t)168296448);             // 128 KB [ROWS][2]
    int*   idxbuf = (int*)  (ws + (size_t)168427520);             // 64 KB  [B][E][K]
    float* wgtbuf = (float*)(ws + (size_t)168493056);             // 64 KB  [B][E][K]
    float* gdot   = (float*)(ws + (size_t)168558592);             // 32 KB  [E][D]
    float* bdot   = (float*)(ws + (size_t)168591360);             // 32 KB  [E][D]
    float* gpart  = (float*)(ws + (size_t)168624128);             // 512 KB [E][D][16]
    float* bpart  = (float*)(ws + (size_t)169148416);             // 512 KB [E][D][16]

    // zero the scatter-add destination (results); logits fully overwritten
    hipMemsetAsync(results, 0, (size_t)B_NUM * S_NUM * D_DIM * sizeof(float), stream);

    // 1) convert inputs to bf16
    cvt_inputs_kernel<<<(B_NUM * S_NUM * D_DIM) / (256 * 8), 256, 0, stream>>>(inputs, Xbf);
    // 2) W1 [E][D][F] -> W1T [E][F][D];  W2 [E][F][D] --(x gamma[f])--> W2gT [E][D][F]
    transpose_cvt_kernel<<<dim3(F_DIM / 32, D_DIM / 32, E_NUM), dim3(32, 8), 0, stream>>>(
        W1, W1T, D_DIM, F_DIM, (const float*)nullptr);
    transpose_cvt_kernel<<<dim3(D_DIM / 32, F_DIM / 32, E_NUM), dim3(32, 8), 0, stream>>>(
        W2, W2gT, F_DIM, D_DIM, gamma);
    // 3) gdot/bdot precompute (deterministic two-pass)
    w2dot_partial_kernel<<<dim3(D_DIM / 256, 16, E_NUM), 256, 0, stream>>>(
        W2, gamma, beta, gpart, bpart);
    w2dot_reduce_kernel<<<(E_NUM * D_DIM) / 256, 256, 0, stream>>>(gpart, bpart, gdot, bdot);
    // 4) router + softmax
    router_kernel<<<(B_NUM * S_NUM) / 8, 256, 0, stream>>>(inputs, Wr, logits, probsT);
    // 5) expert-choice top-k
    topk_kernel<<<B_NUM * E_NUM, 512, 0, stream>>>(probsT, idxbuf, wgtbuf);
    // 6) fc1 + GELU
    fc1_kernel<<<dim3(F_DIM / 128, ROWS / 128), 256, 0, stream>>>(Xbf, W1T, b1, idxbuf, hbuf);
    // 7) LN stats
    ln_stats_kernel<<<ROWS / 8, 256, 0, stream>>>(hbuf, stats);
    // 8) fc2 + weighted scatter
    fc2_kernel<<<dim3(D_DIM / 128, ROWS / 128), 256, 0, stream>>>(
        hbuf, W2gT, gdot, bdot, b2, stats, idxbuf, wgtbuf, results);
}
